// GambaCell_21328807592166
// MI455X (gfx1250) — matrix operand order repaired and checked by reading, not compiled, not run on hardware
//
#include <hip/hip_runtime.h>
#include <stdint.h>

// ---------------------------------------------------------------------------
// Gamba cell (Mamba block on 32x32x256 feature maps), CDNA5 / gfx1250.
// GEMMs: v_wmma_f32_16x16x32_bf16, register double-buffered, 4 waves/block.
// Scan: one workgroup per batch; B_t/C_t staged via global_load_async_to_lds.
// ---------------------------------------------------------------------------

#define B_  16
#define C_  256
#define H_  32
#define W_  32
#define L_  1024            // H_*W_
#define LP_ 1026            // L_ + 2 pad rows (for k=3 gate conv)
#define DI_ 512             // d_inner
#define DS_ 64              // d_state
#define DTR_ 16             // dt_rank
#define M_  (B_*L_)         // 16384 GEMM rows
#define XDBL_ 144           // dt_rank + 2*d_state
#define GWAVES 4            // waves per GEMM block

typedef __attribute__((ext_vector_type(16))) __bf16 v16bf;
typedef __attribute__((ext_vector_type(8)))  float  v8f;

union FragBF { v16bf v; uint4 q[2]; };

static __device__ __forceinline__ uint16_t f2bf(float f) {
    union { float f; uint32_t u; } a; a.f = f;
    uint32_t u = a.u;
    uint32_t r = (u + 0x7FFFu + ((u >> 16) & 1u)) >> 16;   // RNE
    return (uint16_t)r;
}
static __device__ __forceinline__ float sigmoidf_(float x) {
    return 1.0f / (1.0f + __expf(-x));
}

// ---------------------------------------------------------------------------
// Prep: xs_pad[b, lp, c] (bf16, padded sequence) = x + rel_w + rel_h, 0 on pads
// ---------------------------------------------------------------------------
__global__ void k_prep(const float* __restrict__ x, const float* __restrict__ rel_h,
                       const float* __restrict__ rel_w, uint16_t* __restrict__ xs_pad) {
    int idx = blockIdx.x * blockDim.x + threadIdx.x;
    if (idx >= B_ * LP_ * C_) return;
    int c    = idx % C_;
    int rest = idx / C_;
    int lp   = rest % LP_;
    int b    = rest / LP_;
    float v = 0.0f;
    if (lp > 0 && lp < LP_ - 1) {
        int l = lp - 1;
        int p = l >> 5, q = l & 31;
        v = x[((size_t)b * C_ + c) * L_ + l] + rel_w[c * 32 + p] + rel_h[c * 32 + q];
    }
    xs_pad[(size_t)(b * LP_ + lp) * C_ + c] = f2bf(v);
}

// ---------------------------------------------------------------------------
// Weight packing into WMMA B-fragment order:
//   element index = ((kstep*ntiles + ntile)*32 + lane)*16 + j
//   k = kstep*32 + (lane>=16 ? 16 : 0) + j ; n = ntile*16 + (lane&15)
// ---------------------------------------------------------------------------
__global__ void k_pack_generic(const float* __restrict__ src, uint16_t* __restrict__ dst,
                               int K, int N) {
    int idx = blockIdx.x * blockDim.x + threadIdx.x;
    if (idx >= K * N) return;
    int j     = idx & 15;
    int lane  = (idx >> 4) & 31;
    int rest  = idx >> 9;
    int ntil  = rest % (N >> 4);
    int kstep = rest / (N >> 4);
    int k = (kstep << 5) + ((lane >= 16) ? 16 : 0) + j;
    int n = (ntil << 4) + (lane & 15);
    dst[idx] = f2bf(src[(size_t)n * K + k]);
}

// gate_w is (O=256, I=256, K=3); effective GEMM weight (k,n) with k = shift*256+i
__global__ void k_pack_gate(const float* __restrict__ gate_w, uint16_t* __restrict__ dst) {
    const int K = 768, N = 256;
    int idx = blockIdx.x * blockDim.x + threadIdx.x;
    if (idx >= K * N) return;
    int j     = idx & 15;
    int lane  = (idx >> 4) & 31;
    int rest  = idx >> 9;
    int ntil  = rest % (N >> 4);
    int kstep = rest / (N >> 4);
    int k = (kstep << 5) + ((lane >= 16) ? 16 : 0) + j;
    int n = (ntil << 4) + (lane & 15);
    int shift = k / 256, i = k % 256;
    dst[idx] = f2bf(gate_w[(size_t)n * 768 + i * 3 + shift]);
}

__global__ void k_aneg(const float* __restrict__ A_log, float* __restrict__ Aneg) {
    int idx = blockIdx.x * blockDim.x + threadIdx.x;
    if (idx >= DI_ * DS_) return;
    Aneg[idx] = -__expf(A_log[idx]);
}

// ---------------------------------------------------------------------------
// Generic WMMA GEMM: 4 waves/block; wave w owns M-strip (blockIdx.x*4+w)*16,
// all waves stream the same B fragments (WGP$ hits). Register double-buffered:
// k+1 fragments are fetched into fresh registers before the k-step WMMAs so
// loads overlap matrix ops instead of serializing on s_wait_loadcnt 0.
//   A: bf16 row-major, row m -> A + (m/Lrows)*batchStride + (m%Lrows)*lda
//   Bp: fragment-packed bf16, C: fp32 row-major (ldc)
// ep: 0 = plain, 1 = sigmoid(v + bias[col]), 2 = v * mul[row*ldc+col]
// ---------------------------------------------------------------------------
template <int NT>
__global__ void __launch_bounds__(GWAVES * 32)
k_wmma_gemm(const uint16_t* __restrict__ A, const uint16_t* __restrict__ Bp,
            float* __restrict__ Cout, int Lrows, int batchStride, int lda,
            int K, int ntilesTot, int ldc, int ep,
            const float* __restrict__ bias, const float* __restrict__ mul) {
    const int lane    = threadIdx.x & 31;
    const int wave    = threadIdx.x >> 5;
    const int laneLow = lane & 15;
    const int laneHi  = lane >> 4;
    const int m0      = (blockIdx.x * GWAVES + wave) << 4;
    const int nt0     = blockIdx.y * NT;

    const int b  = m0 / Lrows;
    const int l0 = m0 % Lrows;
    const uint16_t* arow = A + (size_t)b * batchStride + (size_t)(l0 + laneLow) * lda;

    auto loadA = [&](int ks, FragBF& f) {
        // A fragment (16x32 bf16): lanes 0-15 hold K 0-15, lanes 16-31 K 16-31
        const int kb = (ks << 5) + (laneHi ? 16 : 0);
        f.q[0] = *reinterpret_cast<const uint4*>(arow + kb);
        f.q[1] = *reinterpret_cast<const uint4*>(arow + kb + 8);
    };
    auto loadB = [&](int ks, int ntg, FragBF& f) {
        const uint4* bp = reinterpret_cast<const uint4*>(
            Bp + ((size_t)ks * ntilesTot + ntg) * 512 + lane * 16);
        f.q[0] = bp[0];
        f.q[1] = bp[1];
    };

    v8f acc[NT];
    const v8f vz = {0.f, 0.f, 0.f, 0.f, 0.f, 0.f, 0.f, 0.f};
#pragma unroll
    for (int nt = 0; nt < NT; ++nt) acc[nt] = vz;

    const int ksteps = K >> 5;

    FragBF fa, fb[NT];
    loadA(0, fa);
#pragma unroll
    for (int nt = 0; nt < NT; ++nt) loadB(0, nt0 + nt, fb[nt]);

    for (int ks = 0; ks < ksteps - 1; ++ks) {
        FragBF fa2, fb2[NT];
        loadA(ks + 1, fa2);
#pragma unroll
        for (int nt = 0; nt < NT; ++nt) loadB(ks + 1, nt0 + nt, fb2[nt]);
        __builtin_prefetch(arow + ((ks + 2) << 5), 0, 3);   // global_prefetch_b8
#pragma unroll
        for (int nt = 0; nt < NT; ++nt)
            acc[nt] = __builtin_amdgcn_wmma_f32_16x16x32_bf16(
                false, fa.v, false, fb[nt].v, (short)0, acc[nt], false, false);
        fa = fa2;
#pragma unroll
        for (int nt = 0; nt < NT; ++nt) fb[nt] = fb2[nt];
    }
#pragma unroll
    for (int nt = 0; nt < NT; ++nt)
        acc[nt] = __builtin_amdgcn_wmma_f32_16x16x32_bf16(
            false, fa.v, false, fb[nt].v, (short)0, acc[nt], false, false);

#pragma unroll
    for (int nt = 0; nt < NT; ++nt) {
        const int col = ((nt0 + nt) << 4) + laneLow;
#pragma unroll
        for (int j = 0; j < 8; ++j) {
            const int row = m0 + (laneHi << 3) + j;
            float v = acc[nt][j];
            if (ep == 1)      v = sigmoidf_(v + bias[col]);
            else if (ep == 2) v = v * mul[(size_t)row * ldc + col];
            Cout[(size_t)row * ldc + col] = v;
        }
    }
}

// ---------------------------------------------------------------------------
// Depthwise causal conv (k=4) + SiLU; emits fp32 u (for scan) and bf16 u (GEMM)
// ---------------------------------------------------------------------------
__global__ void k_conv(const float* __restrict__ xz, const float* __restrict__ conv_w,
                       const float* __restrict__ conv_b, float* __restrict__ u,
                       uint16_t* __restrict__ u_bf) {
    int idx = blockIdx.x * blockDim.x + threadIdx.x;
    if (idx >= M_ * DI_) return;
    int d = idx & (DI_ - 1);
    int m = idx >> 9;
    int b = m >> 10, l = m & (L_ - 1);
    float acc = conv_b[d];
#pragma unroll
    for (int k = 0; k < 4; ++k) {
        int ll = l - 3 + k;
        if (ll >= 0)
            acc += conv_w[d * 4 + k] * xz[((size_t)(b * L_ + ll)) * (2 * DI_) + d];
    }
    float s = acc * sigmoidf_(acc);     // silu
    u[idx]    = s;
    u_bf[idx] = f2bf(s);
}

// ---------------------------------------------------------------------------
// dt_proj (K=16) + softplus -> delta
// ---------------------------------------------------------------------------
__global__ void k_dt(const float* __restrict__ x_dbl, const float* __restrict__ dt_w,
                     const float* __restrict__ dt_b, float* __restrict__ delta) {
    int idx = blockIdx.x * blockDim.x + threadIdx.x;
    if (idx >= M_ * DI_) return;
    int d = idx & (DI_ - 1);
    int m = idx >> 9;
    const float* xd = x_dbl + (size_t)m * XDBL_;
    float acc = dt_b[d];
#pragma unroll
    for (int r = 0; r < DTR_; ++r) acc += xd[r] * dt_w[d * DTR_ + r];
    delta[idx] = (acc > 20.0f) ? acc : log1pf(__expf(acc));
}

// ---------------------------------------------------------------------------
// Selective scan: one workgroup per batch (1024 thr = 32 waves).
// thread t -> channel d = t>>1, state half = (t&1)*32 ; h[32], A[32] in VGPRs.
// B_t/C_t (128 contiguous floats) staged to LDS by wave 0 with a single
// wave-wide GLOBAL_LOAD_ASYNC_TO_LDS_B128 (ASYNCcnt path), then broadcast.
// Half-sums combined with a wave32 shfl_xor.
// ---------------------------------------------------------------------------
__global__ void __launch_bounds__(1024)
k_scan(const float* __restrict__ delta, const float* __restrict__ u,
       const float* __restrict__ x_dbl, const float* __restrict__ Aneg,
       const float* __restrict__ Dvec, float* __restrict__ yacc) {
    const int b    = blockIdx.x;
    const int tid  = threadIdx.x;
    const int d    = tid >> 1;
    const int half = tid & 1;
    const int n0   = half << 5;

    __shared__ float sBC[2 * DS_];          // [0:64) = B_t, [64:128) = C_t

    float Ar[32];
    float h[32];
#pragma unroll
    for (int i = 0; i < 32; ++i) {
        Ar[i] = Aneg[(size_t)d * DS_ + n0 + i];
        h[i]  = 0.0f;
    }
    const float Dd = Dvec[d];

    for (int l = 0; l < L_; ++l) {
        const size_t m = (size_t)b * L_ + l;
        if (tid < 32) {
            // lane t copies 16 B of x_dbl[m, 16 + 4t .. 16 + 4t+3] into LDS
            uint64_t gaddr = (uint64_t)(uintptr_t)(x_dbl + m * XDBL_ + DTR_ + tid * 4);
            uint32_t laddr = (uint32_t)(uintptr_t)(&sBC[tid * 4]);
            asm volatile("global_load_async_to_lds_b128 %0, %1, off"
                         :: "v"(laddr), "v"(gaddr) : "memory");
            asm volatile("s_wait_asynccnt 0x0" ::: "memory");
        }
        __syncthreads();

        const float dv = delta[m * DI_ + d];
        const float uv = u[m * DI_ + d];
        const float du = dv * uv;
        float y = 0.0f;
#pragma unroll
        for (int i = 0; i < 32; ++i) {
            const int n  = n0 + i;
            const float dA = __expf(dv * Ar[i]);
            h[i] = dA * h[i] + du * sBC[n];
            y    = fmaf(h[i], sBC[DS_ + n], y);
        }
        y += __shfl_xor(y, 1, 32);              // combine the two state halves
        if (!half) yacc[m * DI_ + d] = y + uv * Dd;
        __syncthreads();
    }
}

// ---------------------------------------------------------------------------
// y = yacc * silu(z), emitted as bf16 for the out_proj WMMA GEMM
// ---------------------------------------------------------------------------
__global__ void k_ymul(const float* __restrict__ yacc, const float* __restrict__ xz,
                       uint16_t* __restrict__ y_bf) {
    int idx = blockIdx.x * blockDim.x + threadIdx.x;
    if (idx >= M_ * DI_) return;
    int d = idx & (DI_ - 1);
    int m = idx >> 9;
    float z = xz[(size_t)m * (2 * DI_) + DI_ + d];
    float v = yacc[idx] * (z * sigmoidf_(z));
    y_bf[idx] = f2bf(v);
}

// ---------------------------------------------------------------------------
// Host-side orchestration
// ---------------------------------------------------------------------------
static inline size_t align256(size_t x) { return (x + 255) & ~(size_t)255; }

extern "C" void kernel_launch(void* const* d_in, const int* in_sizes, int n_in,
                              void* d_out, int out_size, void* d_ws, size_t ws_size,
                              hipStream_t stream) {
    (void)in_sizes; (void)n_in; (void)out_size; (void)ws_size;

    const float* x         = (const float*)d_in[0];
    const float* rel_h     = (const float*)d_in[1];
    const float* rel_w     = (const float*)d_in[2];
    const float* gate_w    = (const float*)d_in[3];
    const float* gate_b    = (const float*)d_in[4];
    const float* in_proj_w = (const float*)d_in[5];
    const float* conv_w    = (const float*)d_in[6];
    const float* conv_b    = (const float*)d_in[7];
    const float* x_proj_w  = (const float*)d_in[8];
    const float* dt_proj_w = (const float*)d_in[9];
    const float* dt_proj_b = (const float*)d_in[10];
    const float* A_log     = (const float*)d_in[11];
    const float* Dvec      = (const float*)d_in[12];
    const float* out_proj_w= (const float*)d_in[13];
    float* out             = (float*)d_out;

    char* ws = (char*)d_ws;
    size_t off = 0;
    auto carve = [&](size_t bytes) { size_t o = off; off = align256(off + bytes); return o; };

    uint16_t* xs_pad = (uint16_t*)(ws + carve((size_t)B_ * LP_ * C_ * 2));
    uint16_t* gateBp = (uint16_t*)(ws + carve((size_t)768 * 256 * 2));
    uint16_t* ipBp   = (uint16_t*)(ws + carve((size_t)256 * 1024 * 2));
    uint16_t* xpBp   = (uint16_t*)(ws + carve((size_t)512 * XDBL_ * 2));
    uint16_t* opBp   = (uint16_t*)(ws + carve((size_t)512 * 256 * 2));
    float*    Aneg   = (float*)   (ws + carve((size_t)DI_ * DS_ * 4));
    float*    ctx    = (float*)   (ws + carve((size_t)M_ * C_ * 4));
    float*    xz     = (float*)   (ws + carve((size_t)M_ * 2 * DI_ * 4));
    float*    u      = (float*)   (ws + carve((size_t)M_ * DI_ * 4));
    uint16_t* u_bf   = (uint16_t*)(ws + carve((size_t)M_ * DI_ * 2));
    float*    x_dbl  = (float*)   (ws + carve((size_t)M_ * XDBL_ * 4));
    float*    delta  = (float*)   (ws + carve((size_t)M_ * DI_ * 4));
    float*    yacc   = (float*)   (ws + carve((size_t)M_ * DI_ * 4));
    uint16_t* y_bf   = (uint16_t*)(ws + carve((size_t)M_ * DI_ * 2));

    const int TB = 256;
    auto nb = [&](int n) { return (n + TB - 1) / TB; };

    // --- prep + packs ---
    k_prep<<<nb(B_ * LP_ * C_), TB, 0, stream>>>(x, rel_h, rel_w, xs_pad);
    k_pack_gate<<<nb(768 * 256), TB, 0, stream>>>(gate_w, gateBp);
    k_pack_generic<<<nb(256 * 1024), TB, 0, stream>>>(in_proj_w, ipBp, 256, 1024);
    k_pack_generic<<<nb(512 * XDBL_), TB, 0, stream>>>(x_proj_w, xpBp, 512, XDBL_);
    k_pack_generic<<<nb(512 * 256), TB, 0, stream>>>(out_proj_w, opBp, 512, 256);
    k_aneg<<<nb(DI_ * DS_), TB, 0, stream>>>(A_log, Aneg);

    const int gx = (M_ / 16) / GWAVES;     // 256 blocks of 4 waves
    const int gthreads = GWAVES * 32;

    // --- gate conv-as-GEMM (K=768 spans rows l-1..l+1 of padded buffer) ---
    {
        dim3 grid(gx, (C_ / 16) / 4);
        k_wmma_gemm<4><<<grid, gthreads, 0, stream>>>(xs_pad, gateBp, ctx,
            L_, LP_ * C_, C_, 768, C_ / 16, C_, /*ep=*/1, gate_b, nullptr);
    }
    // --- in_proj: xz = xs @ W^T  (A base offset +C_ skips the leading pad row)
    {
        dim3 grid(gx, (1024 / 16) / 4);
        k_wmma_gemm<4><<<grid, gthreads, 0, stream>>>(xs_pad + C_, ipBp, xz,
            L_, LP_ * C_, C_, 256, 1024 / 16, 1024, /*ep=*/0, nullptr, nullptr);
    }
    // --- depthwise conv + silu ---
    k_conv<<<nb(M_ * DI_), TB, 0, stream>>>(xz, conv_w, conv_b, u, u_bf);

    // --- x_proj: x_dbl = u @ W^T  (N=144 -> 9 tiles, NT=3) ---
    {
        dim3 grid(gx, (XDBL_ / 16) / 3);
        k_wmma_gemm<3><<<grid, gthreads, 0, stream>>>(u_bf, xpBp, x_dbl,
            L_, L_ * DI_, DI_, 512, XDBL_ / 16, XDBL_, /*ep=*/0, nullptr, nullptr);
    }
    // --- delta = softplus(dt @ W^T + b) ---
    k_dt<<<nb(M_ * DI_), TB, 0, stream>>>(x_dbl, dt_proj_w, dt_proj_b, delta);

    // --- selective scan: one workgroup per batch ---
    k_scan<<<B_, 1024, 0, stream>>>(delta, u, x_dbl, Aneg, Dvec, yacc);

    // --- y *= silu(z) ---
    k_ymul<<<nb(M_ * DI_), TB, 0, stream>>>(yacc, xz, y_bf);

    // --- out_proj + multiply by ctx, write d_out (flat (B,L,C) == (B,C,H,W) view)
    {
        dim3 grid(gx, (C_ / 16) / 4);
        k_wmma_gemm<4><<<grid, gthreads, 0, stream>>>(y_bf, opBp, out,
            L_, L_ * DI_, DI_, 512, C_ / 16, C_, /*ep=*/2, nullptr, ctx);
    }
}
